// MultiScaleAtten_43233140801765
// MI455X (gfx1250) — matrix operand order repaired and checked by reading, not compiled, not run on hardware
//
#include <hip/hip_runtime.h>
#include <hip/hip_bf16.h>

typedef _Float16 f16;
typedef __attribute__((ext_vector_type(16))) _Float16 v16h;
typedef __attribute__((ext_vector_type(8)))  _Float16 v8h;
typedef __attribute__((ext_vector_type(8)))  float    v8f;

// ---------------------------------------------------------------------------
// Fragment loaders matching CDNA5 WMMA 16-bit VGPR layouts (CDNA5 ISA)
// A (16x32, row-major in memory): lane<16 row=M holds K{0..7,16..23},
//                                 lane>=16 holds K{8..15,24..31}.
// B (32x16) loaded from a transposed (N x K) buffer: lane<16 col=N holds
//                                 K{0..7,16..23}, lane>=16 holds K{8..15,24..31}.
// ---------------------------------------------------------------------------
__device__ __forceinline__ v16h load_a16(const f16* A, int lda, int m0, int k0, int lane) {
  const int row  = m0 + (lane & 15);
  const int ksel = (lane >> 4) << 3;           // 0 or 8
  const f16* p = A + (size_t)row * lda + k0 + ksel;
  v8h lo = *(const v8h*)(p);                   // K = ksel .. ksel+7
  v8h hi = *(const v8h*)(p + 16);              // K = ksel+16 .. ksel+23
  v16h a;
#pragma unroll
  for (int i = 0; i < 8; ++i) { a[i] = lo[i]; a[i + 8] = hi[i]; }
  return a;
}

__device__ __forceinline__ v16h load_bT16(const f16* BT, int ldb, int n0, int k0, int lane) {
  const int col  = n0 + (lane & 15);
  const int ksel = (lane >> 4) << 3;           // 0 or 8
  const f16* p = BT + (size_t)col * ldb + k0 + ksel;
  v8h lo = *(const v8h*)(p);
  v8h hi = *(const v8h*)(p + 16);
  v16h b;
#pragma unroll
  for (int i = 0; i < 8; ++i) { b[i] = lo[i]; b[i + 8] = hi[i]; }
  return b;
}

#define WMMA_F16(a, b, c) \
  __builtin_amdgcn_wmma_f32_16x16x32_f16(false, (a), false, (b), (short)0, (c), false, false)

// ---------------------------------------------------------------------------
// Conversion / transpose helpers
// ---------------------------------------------------------------------------
__global__ void cvt_f32_to_f16_kernel(const float* __restrict__ src, f16* __restrict__ dst, int n) {
  int i = blockIdx.x * 256 + threadIdx.x;
  if (i < n) dst[i] = (f16)src[i];
}

// src: K x N row-major f32  ->  dst: N x K row-major f16 (i.e. B^T)
__global__ void transpose_cvt_kernel(const float* __restrict__ src, f16* __restrict__ dst,
                                     int K, int N) {
  int i = blockIdx.x * 256 + threadIdx.x;
  if (i < K * N) {
    int k = i / N, n = i % N;
    dst[(size_t)n * K + k] = (f16)src[i];
  }
}

// ---------------------------------------------------------------------------
// Generic wave-tiled WMMA GEMM: C[M,N] = A[M,K](f16) * BT[N,K]^T(f16) + bias
// One wave computes a 64x64 tile (4x4 grid of 16x16x32 WMMAs).
// ---------------------------------------------------------------------------
template <bool STORE_F16>
__global__ __launch_bounds__(128) void wmma_gemm_kernel(
    const f16* __restrict__ A, const f16* __restrict__ BT,
    const float* __restrict__ bias, void* __restrict__ C,
    int M, int N, int K) {
  const int lane = threadIdx.x & 31;
  const int wave = threadIdx.x >> 5;
  const int mtiles = M >> 6;
  const int ntiles = N >> 6;
  const int tile = blockIdx.x * 4 + wave;
  if (tile >= mtiles * ntiles) return;
  const int m0 = (tile % mtiles) << 6;   // M-fast for B reuse in L2
  const int n0 = (tile / mtiles) << 6;

  v8f acc[4][4] = {};
  for (int k0 = 0; k0 < K; k0 += 32) {
    v16h b[4];
#pragma unroll
    for (int nn = 0; nn < 4; ++nn) b[nn] = load_bT16(BT, K, n0 + nn * 16, k0, lane);
#pragma unroll
    for (int mm = 0; mm < 4; ++mm) {
      v16h a = load_a16(A, K, m0 + mm * 16, k0, lane);
#pragma unroll
      for (int nn = 0; nn < 4; ++nn) acc[mm][nn] = WMMA_F16(a, b[nn], acc[mm][nn]);
    }
  }

  // C/D layout: VGPR r -> row m0+r (+8 for lanes>=16), lane&15 -> column.
#pragma unroll
  for (int nn = 0; nn < 4; ++nn) {
    const int col = n0 + nn * 16 + (lane & 15);
    const float bv = bias[col];
#pragma unroll
    for (int mm = 0; mm < 4; ++mm) {
#pragma unroll
      for (int r = 0; r < 8; ++r) {
        const int row = m0 + mm * 16 + r + ((lane >> 4) << 3);
        const float v = acc[mm][nn][r] + bv;
        if (STORE_F16) ((f16*)C)[(size_t)row * N + col] = (f16)v;
        else           ((float*)C)[(size_t)row * N + col] = v;
      }
    }
  }
}

// ---------------------------------------------------------------------------
// Per-(block, head) attention: S = Q K^T ; P = softmax(S) ; O = P V
// qkv: [32768 tokens][1536] f16, column = which*512 + h*64 + d
// 128 threads = 4 waves; wave w owns 16-wide column strip.
// ---------------------------------------------------------------------------
__global__ __launch_bounds__(128) void attn_kernel(const f16* __restrict__ qkv,
                                                   f16* __restrict__ attnout) {
  const int blk  = blockIdx.x >> 3;
  const int h    = blockIdx.x & 7;
  const int tid  = threadIdx.x;
  const int lane = tid & 31;
  const int wave = tid >> 5;
  const int LDQ  = 1536;
  const int LDP  = 80;   // padded f16 LDS stride (160 B: keeps 16B alignment)

  __shared__ float Slds[64 * 64];
  __shared__ f16   Plds[64 * LDP];
  __shared__ f16   VTlds[64 * LDP];

  const f16* Qb = qkv + (size_t)blk * 64 * LDQ + 0 * 512 + h * 64;
  const f16* Kb = qkv + (size_t)blk * 64 * LDQ + 1 * 512 + h * 64;
  const f16* Vb = qkv + (size_t)blk * 64 * LDQ + 2 * 512 + h * 64;

  // Stage V^T into LDS: VT[d][t] = V[t][d]
  for (int idx = tid; idx < 64 * 64; idx += 128) {
    int t = idx >> 6, d = idx & 63;
    VTlds[d * LDP + t] = Vb[(size_t)t * LDQ + d];
  }
  __syncthreads();

  // S = Q K^T : wave computes key-columns m0 = wave*16, all 64 query rows.
  {
    const int n0 = wave * 16;
    v8f acc[4] = {};
    for (int k0 = 0; k0 < 64; k0 += 32) {
      v16h b = load_bT16(Kb, LDQ, n0, k0, lane);   // rows of K are B^T rows
#pragma unroll
      for (int mm = 0; mm < 4; ++mm) {
        v16h a = load_a16(Qb, LDQ, mm * 16, k0, lane);
        acc[mm] = WMMA_F16(a, b, acc[mm]);
      }
    }
#pragma unroll
    for (int mm = 0; mm < 4; ++mm)
#pragma unroll
      for (int r = 0; r < 8; ++r) {
        int row = mm * 16 + r + ((lane >> 4) << 3);
        int col = n0 + (lane & 15);
        Slds[row * 64 + col] = acc[mm][r];
      }
  }
  __syncthreads();

  // Row softmax (reference applies no scale), P stored as f16.
  if (tid < 64) {
    float mx = -1e30f;
    for (int m = 0; m < 64; ++m) mx = fmaxf(mx, Slds[tid * 64 + m]);
    float sum = 0.f;
    for (int m = 0; m < 64; ++m) {
      float e = __expf(Slds[tid * 64 + m] - mx);
      sum += e;
      Slds[tid * 64 + m] = e;
    }
    float inv = 1.0f / sum;
    for (int m = 0; m < 64; ++m) Plds[tid * LDP + m] = (f16)(Slds[tid * 64 + m] * inv);
  }
  __syncthreads();

  // O = P V : wave computes value-dim columns d0 = wave*16.
  {
    const int n0 = wave * 16;
    v8f acc[4] = {};
    for (int k0 = 0; k0 < 64; k0 += 32) {
      v16h b = load_bT16(VTlds, LDP, n0, k0, lane);
#pragma unroll
      for (int mm = 0; mm < 4; ++mm) {
        v16h a = load_a16(Plds, LDP, mm * 16, k0, lane);
        acc[mm] = WMMA_F16(a, b, acc[mm]);
      }
    }
#pragma unroll
    for (int mm = 0; mm < 4; ++mm)
#pragma unroll
      for (int r = 0; r < 8; ++r) {
        int row = mm * 16 + r + ((lane >> 4) << 3);
        int col = n0 + (lane & 15);
        attnout[((size_t)blk * 64 + row) * 512 + h * 64 + col] = (f16)acc[mm][r];
      }
  }
}

// ---------------------------------------------------------------------------
extern "C" void kernel_launch(void* const* d_in, const int* in_sizes, int n_in,
                              void* d_out, int out_size, void* d_ws, size_t ws_size,
                              hipStream_t stream) {
  const float* x      = (const float*)d_in[0];  // (8,8,8,64,512)
  const float* w_qkv  = (const float*)d_in[1];  // (512,1536)
  const float* b_qkv  = (const float*)d_in[2];  // (1536)
  const float* w_proj = (const float*)d_in[3];  // (512,512)
  const float* b_proj = (const float*)d_in[4];  // (512)
  float* out = (float*)d_out;

  const int M = 8 * 8 * 8 * 64;  // 32768 tokens
  const int C = 512;
  const int NQ = 3 * C;          // 1536

  // Workspace carve-up (f16 staging, all 256B aligned)
  char* ws = (char*)d_ws;
  f16* xh     = (f16*)ws;  ws += (size_t)M * C * sizeof(f16);       // 32 MB
  f16* qkvh   = (f16*)ws;  ws += (size_t)M * NQ * sizeof(f16);      // 96 MB
  f16* attnh  = (f16*)ws;  ws += (size_t)M * C * sizeof(f16);       // 32 MB
  f16* wqkvT  = (f16*)ws;  ws += (size_t)NQ * C * sizeof(f16);      // 1.5 MB
  f16* wprojT = (f16*)ws;  ws += (size_t)C * C * sizeof(f16);       // 0.5 MB

  // 1) f32 -> f16 conversions / weight transposes
  {
    int n = M * C;
    cvt_f32_to_f16_kernel<<<(n + 255) / 256, 256, 0, stream>>>(x, xh, n);
    n = C * NQ;
    transpose_cvt_kernel<<<(n + 255) / 256, 256, 0, stream>>>(w_qkv, wqkvT, C, NQ);
    n = C * C;
    transpose_cvt_kernel<<<(n + 255) / 256, 256, 0, stream>>>(w_proj, wprojT, C, C);
  }

  // 2) QKV GEMM: qkvh = xh @ w_qkv + b_qkv   (f16 out)
  {
    int tiles = (M / 64) * (NQ / 64);  // 12288
    wmma_gemm_kernel<true><<<tiles / 4, 128, 0, stream>>>(xh, wqkvT, b_qkv, qkvh, M, NQ, C);
  }

  // 3) Per-(block, head) attention
  attn_kernel<<<512 * 8, 128, 0, stream>>>(qkvh, attnh);

  // 4) Projection GEMM: out = attnh @ w_proj + b_proj  (f32 out)
  {
    int tiles = (M / 64) * (C / 64);   // 4096
    wmma_gemm_kernel<false><<<tiles / 4, 128, 0, stream>>>(attnh, wprojT, b_proj, out, M, C, C);
  }
}
